// SymmetryEnhancedAttention_49022756716608
// MI455X (gfx1250) — matrix operand order repaired and checked by reading, not compiled, not run on hardware
//
#include <hip/hip_runtime.h>
#include <hip/hip_bf16.h>

// ---------- problem constants ----------
#define Sn 3
#define Bn 2
#define Cn 256
#define Hn 64
#define Wn 64
#define Pn 2
#define Qn 2
#define Dn 128          // C/2
#define HPn 32
#define WPn 32
#define Nn 1024         // tokens per partition
#define HW 4096

typedef __bf16 v16bf __attribute__((ext_vector_type(16)));
typedef float  v8f   __attribute__((ext_vector_type(8)));
typedef int    v4i   __attribute__((ext_vector_type(4)));
typedef int    v2i   __attribute__((ext_vector_type(2)));

typedef __attribute__((address_space(1))) v4i* gv4p;
typedef __attribute__((address_space(3))) v4i* lv4p;
typedef __attribute__((address_space(1))) v2i* gv2p;
typedef __attribute__((address_space(3))) v2i* lv2p;

#ifndef __has_builtin
#define __has_builtin(x) 0
#endif

#if __has_builtin(__builtin_amdgcn_global_load_async_to_lds_b128) && \
    __has_builtin(__builtin_amdgcn_global_load_async_to_lds_b64)  && \
    __has_builtin(__builtin_amdgcn_s_wait_asynccnt)
#define USE_ASYNC_LDS 1
#endif

// 16-byte global -> LDS copy (async on gfx1250: ASYNCcnt-tracked, no VGPR data)
__device__ __forceinline__ void cp16(void* lds, const void* g) {
#ifdef USE_ASYNC_LDS
    __builtin_amdgcn_global_load_async_to_lds_b128((gv4p)g, (lv4p)lds, 0, 0);
#else
    *(uint4*)lds = *(const uint4*)g;
#endif
}
// 8-byte global -> LDS copy
__device__ __forceinline__ void cp8(void* lds, const void* g) {
#ifdef USE_ASYNC_LDS
    __builtin_amdgcn_global_load_async_to_lds_b64((gv2p)g, (lv2p)lds, 0, 0);
#else
    *(uint2*)lds = *(const uint2*)g;
#endif
}
__device__ __forceinline__ void cp_wait() {
#ifdef USE_ASYNC_LDS
    __builtin_amdgcn_s_wait_asynccnt(0);
#endif
}

// ---- CDNA5 WMMA fragment index helpers (contiguous k = j + 16*(lane/16) for A and B, wave32) ----
__device__ __forceinline__ int a_kidx(int j, int lane) {
    return j + 16 * (lane >> 4);
}
__device__ __forceinline__ int b_kidx(int j, int lane) {
    return j + 16 * (lane >> 4);
}
__device__ __forceinline__ int cd_row(int r, int lane) {
    return r + 8 * (lane >> 4);
}

__device__ __forceinline__ v8f wmma_bf16(v16bf a, v16bf b, v8f c) {
    return __builtin_amdgcn_wmma_f32_16x16x32_bf16(
        false, a, false, b, (short)0, c, false, false);
}

// =====================================================================
// Kernel 0: one-time f32 -> bf16 conversion of all weight matrices.
// Layout in dst: [theta | phi | g | h] (4 x D*C) then out_w (C*C).
// =====================================================================
__global__ __launch_bounds__(256) void wconv_kernel(
    const float* __restrict__ tw, const float* __restrict__ pw,
    const float* __restrict__ gw, const float* __restrict__ hw_,
    const float* __restrict__ ow, __bf16* __restrict__ wdst)
{
    const int i = blockIdx.x * 256 + threadIdx.x;   // 0 .. 196607
    const int WSZ = Dn * Cn;                        // 32768
    float v;
    if      (i <     WSZ) v = tw[i];
    else if (i < 2 * WSZ) v = pw[i -     WSZ];
    else if (i < 3 * WSZ) v = gw[i - 2 * WSZ];
    else if (i < 4 * WSZ) v = hw_[i - 3 * WSZ];
    else                  v = ow[i - 4 * WSZ];
    wdst[i] = (__bf16)v;
}

// =====================================================================
// Kernel 1: four 1x1-conv projections  out[n,d] = sum_c x[c,n]*w[d,c]+b
// Grid: S*B*P*Q * (N/16) = 1536 blocks, 256 threads (8 waves).
// Wave w owns d-tile w (16 outputs), loops the 4 projections.
// B fragments are single contiguous 32-B v16bf loads from bf16 weights.
// =====================================================================
__global__ __launch_bounds__(256) void proj_kernel(
    const float* __restrict__ x,
    const __bf16* __restrict__ wb,                 // [theta|phi|g|h] bf16
    const float* __restrict__ tb, const float* __restrict__ pb,
    const float* __restrict__ gb, const float* __restrict__ hb_,
    __bf16* __restrict__ qbuf, __bf16* __restrict__ kbuf,
    __bf16* __restrict__ vbuf, __bf16* __restrict__ hbuf)
{
    __shared__ float Xf[32][16];   // [c_local][token], staged raw f32

    int bid = blockIdx.x;
    const int tile = bid & 63; bid >>= 6;
    const int qq = bid & 1; bid >>= 1;
    const int p  = bid & 1; bid >>= 1;
    const int b  = bid & 1; bid >>= 1;
    const int s  = bid;                 // 0..2

    const int tid  = threadIdx.x;
    const int lane = tid & 31;
    const int wave = tid >> 5;          // d-tile
    const int n0   = tile * 16;
    const int hp   = n0 >> 5;
    const int wp0  = n0 & 31;
    const int h    = p * 32 + hp;
    const int wb_w = qq * 32 + wp0;

    const float* bsel[4] = { tb, pb, gb, hb_ };

    v8f acc[4];
    #pragma unroll
    for (int pr = 0; pr < 4; ++pr)
        acc[pr] = (v8f){0.f,0.f,0.f,0.f,0.f,0.f,0.f,0.f};

    const size_t xplane = ((size_t)(s * Bn + b) * Cn) * HW + (size_t)h * Wn + wb_w;
    const int d = wave * 16 + (lane & 15);

    for (int ck = 0; ck < 8; ++ck) {            // K chunks of 32 channels
        const int c0 = ck * 32;
        __syncthreads();
        {   // async-stage 32 channels x 16 tokens of f32 (2 KB), 8 B per thread
            const int cl = tid >> 3, pr2 = (tid & 7) * 2;
            cp8(&Xf[cl][pr2], &x[xplane + (size_t)(c0 + cl) * HW + pr2]);
        }
        cp_wait();
        __syncthreads();

        // A fragment: 16 tokens x 32 channels (shared by all 4 projections)
        v16bf afr;
        const int m = lane & 15;
        #pragma unroll
        for (int j = 0; j < 16; ++j) afr[j] = (__bf16)Xf[a_kidx(j, lane)][m];

        #pragma unroll
        for (int pr = 0; pr < 4; ++pr) {
            const v16bf bfr = *(const v16bf*)(wb + (size_t)pr * (Dn * Cn) +
                                              (size_t)d * Cn + c0 + 16 * (lane >> 4));
            acc[pr] = wmma_bf16(afr, bfr, acc[pr]);
        }
    }

    // store results (+bias) as bf16, layout [..][N][D]
    const size_t pqbase  = ((size_t)(b * Pn + p) * Qn + qq) * Nn;
    const size_t spqbase = (((size_t)(s * Bn + b) * Pn + p) * Qn + qq) * Nn;

    #pragma unroll
    for (int pr = 0; pr < 4; ++pr) {
        const float bias = bsel[pr][d];
        #pragma unroll
        for (int r = 0; r < 8; ++r) {
            const int row = n0 + cd_row(r, lane);
            const __bf16 val = (__bf16)(acc[pr][r] + bias);
            if (pr == 0) {
                if (s == 1) qbuf[(pqbase + row) * Dn + d] = val;
            } else if (pr == 1) {
                kbuf[(spqbase + row) * Dn + d] = val;
            } else if (pr == 2) {
                vbuf[(spqbase + row) * Dn + d] = val;
            } else {
                hbuf[(spqbase + row) * Dn + d] = val;
            }
        }
    }
}

// =====================================================================
// Kernel 2: flash-style attention, per-s softmax, summed over s.
// Grid: sym(2)*b(2)*p(2)*q(2)*rowtiles(32) = 512 blocks, 256 threads.
// Block owns 32 query rows; waves: qt = w/4 (row half),
//   score role: m-chunk = w%4; PV role: d-chunks {2*(w%4), 2*(w%4)+1}.
// K/V tiles staged with global_load_async_to_lds_b128 (ASYNCcnt path).
// =====================================================================
__global__ __launch_bounds__(256) void attn_kernel(
    const __bf16* __restrict__ qbuf, const __bf16* __restrict__ kbuf,
    const __bf16* __restrict__ vbuf, const __bf16* __restrict__ hbuf,
    __bf16* __restrict__ obuf)
{
    __shared__ __bf16 Qs[32][Dn];      // 8 KB
    __shared__ __bf16 Ks[64][Dn];      // 16 KB
    __shared__ __bf16 Vs[64][Dn];      // 16 KB
    __shared__ float  Ss[32][64];      // 8 KB scores
    __shared__ __bf16 Ps[32][64];      // 4 KB probabilities
    __shared__ float  mrow[32], lrow[32], arow[32];

    int bid = blockIdx.x;
    const int rt  = bid & 31; bid >>= 5;
    const int qq  = bid & 1;  bid >>= 1;
    const int p   = bid & 1;  bid >>= 1;
    const int b   = bid & 1;  bid >>= 1;
    const int sym = bid;                     // 0 = self, 1 = mirrored
    const int qk  = sym ? (1 - qq) : qq;     // flipped partition for K and V_sym

    const __bf16* vsrc = sym ? hbuf : vbuf;

    const int tid  = threadIdx.x;
    const int lane = tid & 31;
    const int wave = tid >> 5;
    const int qt   = wave >> 2;              // 0/1: which 16-row half
    const int mc   = wave & 3;               // score m-chunk / PV d-pair
    const int n0   = rt * 32;

    const size_t qbase = (((size_t)(b * Pn + p) * Qn) + qq) * Nn;
    const size_t obase = qbase;
    const float scale  = 0.08838834764831845f;   // 1/sqrt(128)

    // async-stage Q tile once: 32x128 bf16 = 8 KB, 16 B per copy
    #pragma unroll
    for (int e = tid; e < 512; e += 256) {
        const int r = e >> 4, d = (e & 15) * 8;
        cp16(&Qs[r][d], &qbuf[(qbase + n0 + r) * Dn + d]);
    }

    v8f accT[2], accS[2];
    #pragma unroll
    for (int t = 0; t < 2; ++t) {
        accT[t] = (v8f){0.f,0.f,0.f,0.f,0.f,0.f,0.f,0.f};
        accS[t] = (v8f){0.f,0.f,0.f,0.f,0.f,0.f,0.f,0.f};
    }

    for (int s = 0; s < Sn; ++s) {
        const size_t kvb = ((((size_t)(s * Bn + b) * Pn + p) * Qn) + qk) * Nn;
        __syncthreads();
        if (tid < 32) { mrow[tid] = -1e30f; lrow[tid] = 0.f; }

        for (int mt = 0; mt < Nn / 64; ++mt) {
            const int m0 = mt * 64;
            __syncthreads();
            // async-stage K/V tiles (64x128 bf16 = 16 KB each), 16 B per copy
            #pragma unroll
            for (int e = tid; e < 1024; e += 256) {
                const int r = e >> 4, d = (e & 15) * 8;
                cp16(&Ks[r][d], &kbuf[(kvb + m0 + r) * Dn + d]);
                cp16(&Vs[r][d], &vsrc[(kvb + m0 + r) * Dn + d]);
            }
            if (mt + 1 < Nn / 64)    // pull next tile toward L2/WGP$ meanwhile
                __builtin_prefetch(&kbuf[(kvb + m0 + 64) * Dn + (tid & 63) * 64], 0, 1);
            cp_wait();
            __syncthreads();

            // ---- scores: 16x16 per wave, K-dim = D = 128 ----
            v8f sc = (v8f){0.f,0.f,0.f,0.f,0.f,0.f,0.f,0.f};
            #pragma unroll
            for (int ck = 0; ck < 4; ++ck) {
                v16bf afr, bfr;
                const int mrow_q = qt * 16 + (lane & 15);
                const int mm     = mc * 16 + (lane & 15);
                #pragma unroll
                for (int j = 0; j < 16; ++j) {
                    afr[j] = Qs[mrow_q][ck * 32 + a_kidx(j, lane)];
                    bfr[j] = Ks[mm][ck * 32 + b_kidx(j, lane)];
                }
                sc = wmma_bf16(afr, bfr, sc);
            }
            #pragma unroll
            for (int r = 0; r < 8; ++r)
                Ss[qt * 16 + cd_row(r, lane)][mc * 16 + (lane & 15)] = sc[r] * scale;
            __syncthreads();

            // ---- online softmax, one thread per row ----
            if (tid < 32) {
                const int row = tid;
                float mx = mrow[row];
                #pragma unroll 4
                for (int j = 0; j < 64; ++j) mx = fmaxf(mx, Ss[row][j]);
                const float al = __expf(mrow[row] - mx);
                float sum = 0.f;
                #pragma unroll 4
                for (int j = 0; j < 64; ++j) {
                    const float pv = __expf(Ss[row][j] - mx);
                    sum += pv;
                    Ps[row][j] = (__bf16)pv;
                }
                mrow[row] = mx;
                lrow[row] = al * lrow[row] + sum;
                arow[row] = al;
            }
            __syncthreads();

            // ---- PV: rescale running acc, then acc += P @ V ----
            #pragma unroll
            for (int t = 0; t < 2; ++t)
                #pragma unroll
                for (int r = 0; r < 8; ++r)
                    accS[t][r] *= arow[qt * 16 + cd_row(r, lane)];
            #pragma unroll
            for (int ck = 0; ck < 2; ++ck) {
                v16bf afr;
                const int prow = qt * 16 + (lane & 15);
                #pragma unroll
                for (int j = 0; j < 16; ++j)
                    afr[j] = Ps[prow][ck * 32 + a_kidx(j, lane)];
                #pragma unroll
                for (int t = 0; t < 2; ++t) {
                    const int d = (mc * 2 + t) * 16 + (lane & 15);
                    v16bf bfr;
                    #pragma unroll
                    for (int j = 0; j < 16; ++j)
                        bfr[j] = Vs[ck * 32 + b_kidx(j, lane)][d];
                    accS[t] = wmma_bf16(afr, bfr, accS[t]);
                }
            }
        }

        // normalize this slice's contribution and fold into total
        #pragma unroll
        for (int r = 0; r < 8; ++r) {
            const float inv = 1.0f / lrow[qt * 16 + cd_row(r, lane)];
            #pragma unroll
            for (int t = 0; t < 2; ++t) {
                accT[t][r] += accS[t][r] * inv;
                accS[t][r] = 0.f;
            }
        }
    }

    // write [b,p,q,n,c] with channel half selected by sym
    #pragma unroll
    for (int t = 0; t < 2; ++t) {
        const int c = sym * Dn + (mc * 2 + t) * 16 + (lane & 15);
        #pragma unroll
        for (int r = 0; r < 8; ++r) {
            const int n = n0 + qt * 16 + cd_row(r, lane);
            obuf[(obase + n) * Cn + c] = (__bf16)accT[t][r];
        }
    }
}

// =====================================================================
// Kernel 3: out = out_w @ attn_out + out_b + residual  ->  [B,C,H,W] f32
// Grid: 8192 tokens / 16 = 512 blocks, 256 threads; wave owns 2 c_out tiles.
// Token tile (16x256 bf16) async-staged to LDS; weights pre-converted bf16.
// =====================================================================
__global__ __launch_bounds__(256) void outproj_kernel(
    const __bf16* __restrict__ obuf,
    const __bf16* __restrict__ owb, const float* __restrict__ ob,
    const float* __restrict__ x, float* __restrict__ out)
{
    __shared__ __bf16 Obs[16][Cn];     // 8 KB token tile

    const int tid  = threadIdx.x;
    const int lane = tid & 31;
    const int wave = tid >> 5;

    const int t0 = blockIdx.x * 16;          // token = b*H*W + h*W + w
    const int b  = t0 / HW;
    const int h  = (t0 % HW) / Wn;
    const int w0 = t0 % Wn;
    const int p  = h >> 5, hp = h & 31;
    const int qq = w0 >> 5, wp0 = w0 & 31;

    const size_t nbase = (((size_t)(b * Pn + p) * Qn) + qq) * Nn + hp * WPn + wp0;

    // async-stage 16 tokens x 256 channels bf16 (8 KB), 16 B per copy
    #pragma unroll
    for (int e = tid; e < 512; e += 256) {
        const int r = e >> 5, d = (e & 31) * 8;
        cp16(&Obs[r][d], &obuf[(nbase + r) * Cn + d]);
    }
    cp_wait();
    __syncthreads();

    v8f acc[2];
    #pragma unroll
    for (int t = 0; t < 2; ++t)
        acc[t] = (v8f){0.f,0.f,0.f,0.f,0.f,0.f,0.f,0.f};

    for (int ck = 0; ck < 8; ++ck) {
        const int c0 = ck * 32;
        v16bf afr;
        const int m = lane & 15;
        #pragma unroll
        for (int j = 0; j < 16; ++j)
            afr[j] = Obs[m][c0 + a_kidx(j, lane)];
        #pragma unroll
        for (int t = 0; t < 2; ++t) {
            const int co = (wave * 2 + t) * 16 + (lane & 15);
            const v16bf bfr = *(const v16bf*)(owb + (size_t)co * Cn + c0 + 16 * (lane >> 4));
            acc[t] = wmma_bf16(afr, bfr, acc[t]);
        }
    }

    #pragma unroll
    for (int t = 0; t < 2; ++t) {
        const int co   = (wave * 2 + t) * 16 + (lane & 15);
        const float bc = ob[co];
        #pragma unroll
        for (int r = 0; r < 8; ++r) {
            const int ww = w0 + cd_row(r, lane);
            const size_t oidx = (((size_t)b * Cn + co) * Hn + h) * Wn + ww;
            const float res = x[((size_t)(1 * Bn + b) * Cn + co) * HW + (size_t)h * Wn + ww];
            out[oidx] = acc[t][r] + bc + res;
        }
    }
}

// =====================================================================
extern "C" void kernel_launch(void* const* d_in, const int* in_sizes, int n_in,
                              void* d_out, int out_size, void* d_ws, size_t ws_size,
                              hipStream_t stream) {
    const float* x  = (const float*)d_in[0];
    const float* tw = (const float*)d_in[1];
    const float* tb = (const float*)d_in[2];
    const float* pw = (const float*)d_in[3];
    const float* pb = (const float*)d_in[4];
    const float* gw = (const float*)d_in[5];
    const float* gb = (const float*)d_in[6];
    const float* hw_ = (const float*)d_in[7];
    const float* hb_ = (const float*)d_in[8];
    const float* ow = (const float*)d_in[9];
    const float* ob = (const float*)d_in[10];
    float* out = (float*)d_out;

    // workspace partitions (bf16 elements)
    __bf16* ws = (__bf16*)d_ws;
    const size_t QSZ = (size_t)Bn * Pn * Qn * Nn * Dn;        // 1 Mi
    const size_t KSZ = (size_t)Sn * QSZ;                      // 3 Mi
    __bf16* qbuf = ws;
    __bf16* kbuf = qbuf + QSZ;
    __bf16* vbuf = kbuf + KSZ;
    __bf16* hbuf = vbuf + KSZ;
    __bf16* obuf = hbuf + KSZ;                                // 2 Mi ([B,P,Q,N,C])
    __bf16* wb   = obuf + (size_t)Bn * Pn * Qn * Nn * Cn;     // weights: 4*D*C + C*C
    __bf16* owb  = wb + 4 * (size_t)Dn * Cn;

    wconv_kernel<<<(4 * Dn * Cn + Cn * Cn) / 256, 256, 0, stream>>>(
        tw, pw, gw, hw_, ow, wb);

    proj_kernel<<<Sn * Bn * Pn * Qn * (Nn / 16), 256, 0, stream>>>(
        x, wb, tb, pb, gb, hb_, qbuf, kbuf, vbuf, hbuf);

    attn_kernel<<<2 * Bn * Pn * Qn * (Nn / 32), 256, 0, stream>>>(
        qbuf, kbuf, vbuf, hbuf, obuf);

    outproj_kernel<<<(Bn * HW) / 16, 256, 0, stream>>>(
        obuf, owb, ob, x, out);
}
